// Encoder_3908420239637
// MI455X (gfx1250) — matrix operand order repaired and checked by reading, not compiled, not run on hardware
//
#include <hip/hip_runtime.h>
#include <hip/hip_bf16.h>

typedef __attribute__((ext_vector_type(16))) _Float16 v16h;
typedef __attribute__((ext_vector_type(8)))  float    v8f;

#define Bq 16
#define Sq 64
#define Hq 1024
#define Cq 16

__device__ __forceinline__ float sigf(float x) {
    return 1.0f / (1.0f + __expf(-x));
}

// ---------------------------------------------------------------------------
// One-time f32 -> f16 conversion (weights), and f16 zero-init (state shadow).
// ---------------------------------------------------------------------------
__global__ void k_cvt(_Float16* __restrict__ dst, const float* __restrict__ src, int n) {
    int i = blockIdx.x * blockDim.x + threadIdx.x;
    if (i < n) dst[i] = (_Float16)src[i];
}

__global__ void k_zero_h(_Float16* __restrict__ p, int n) {
    int i = blockIdx.x * blockDim.x + threadIdx.x;
    if (i < n) p[i] = (_Float16)0.0f;
}

// ---------------------------------------------------------------------------
// Step kernel 1: fused gate/map GEMMs + sigmoid + product + s-reduction.
//   masked[b,s,h] = state_h[b,s,h] * dep[b,t,s]
//   g = sigmoid(masked @ gate_w.T + gate_b); m = masked @ map_w.T + map_b
//   h_in[b,o] = sum_s g[b,s,o] * m[b,s,o]
// Grid: (64 col-blocks, 16 batches), 128 threads (4 waves; wave w = s-tile w).
// Pure f16 loads feeding v_wmma_f32_16x16x32_f16; no conversions in hot loop.
// ---------------------------------------------------------------------------
__global__ __launch_bounds__(128) void k_gatemap(
    const float*    __restrict__ dep,       // [B,S,S] f32
    const _Float16* __restrict__ state_h,   // [B,S,H] f16 shadow state
    const _Float16* __restrict__ gate_w_h,  // [H,H]   f16
    const float*    __restrict__ gate_b,    // [H]
    const _Float16* __restrict__ map_w_h,   // [H,H]   f16
    const float*    __restrict__ map_b,     // [H]
    float*          __restrict__ h_in,      // [B,H]   f32 out
    _Float16*       __restrict__ h_in_h,    // [B,H]   f16 out
    int t)
{
    const int nb   = blockIdx.x;          // output 16-column block
    const int b    = blockIdx.y;          // batch
    const int wave = threadIdx.x >> 5;    // s-tile 0..3
    const int lane = threadIdx.x & 31;
    const int mrow = lane & 15;
    const int hi   = lane >> 4;

    const int s     = wave * 16 + mrow;   // A-row this lane feeds
    const int oglob = nb * 16 + mrow;     // B/C column this lane owns

    const _Float16 hmval =
        (_Float16)dep[(size_t)b * Sq * Sq + (size_t)t * Sq + s];
    const _Float16* srow  = state_h  + ((size_t)b * Sq + s) * Hq;
    const _Float16* gwrow = gate_w_h + (size_t)oglob * Hq;
    const _Float16* mwrow = map_w_h  + (size_t)oglob * Hq;

    v8f cg = {};
    v8f cm = {};

    const int kbaseA = hi * 16;   // A: hi=0 -> K 0..15 ; hi=1 -> K 16..31
    const int kbaseB = hi * 16;   // B: hi=0 -> K 0..15 ; hi=1 -> K 16..31

    for (int kt = 0; kt < Hq / 32; ++kt) {
        const int k0 = kt * 32;
        v16h a, bg, bm;
#pragma unroll
        for (int j = 0; j < 16; ++j) {
            const int ka = kbaseA + j;
            a[j]  = srow[k0 + ka] * hmval;
            const int kb = kbaseB + j;
            bg[j] = gwrow[k0 + kb];
            bm[j] = mwrow[k0 + kb];
        }
        cg = __builtin_amdgcn_wmma_f32_16x16x32_f16(false, a, false, bg,
                                                    (short)0, cg, false, false);
        cm = __builtin_amdgcn_wmma_f32_16x16x32_f16(false, a, false, bm,
                                                    (short)0, cm, false, false);
    }

    // bias is per output column (constant per lane), then sigmoid*map, reduce rows
    const float gb = gate_b[oglob];
    const float mb = map_b[oglob];
    float partial = 0.0f;
#pragma unroll
    for (int r = 0; r < 8; ++r) {
        const float gv = sigf(cg[r] + gb);
        const float mv = cm[r] + mb;
        partial += gv * mv;
    }
    // lanes L and L+16 hold the other 8 rows of the same column
    partial += __shfl_xor(partial, 16, 32);

    __shared__ float red[4][16];
    if (lane < 16) red[wave][mrow] = partial;
    __syncthreads();
    if (threadIdx.x < 16) {
        const float tot = red[0][threadIdx.x] + red[1][threadIdx.x]
                        + red[2][threadIdx.x] + red[3][threadIdx.x];
        const size_t o = (size_t)b * Hq + nb * 16 + threadIdx.x;
        h_in[o]   = tot;
        h_in_h[o] = (_Float16)tot;
    }
}

// ---------------------------------------------------------------------------
// Step kernel 2: gh = h_in @ w_hh.T + b_hh   ([16,1024] x [1024,3072])
// Grid: 192 blocks of 1 wave; M=16 covers all batches in one WMMA tile.
// ---------------------------------------------------------------------------
__global__ __launch_bounds__(32) void k_gh(
    const _Float16* __restrict__ h_in_h,  // [B,H] f16
    const _Float16* __restrict__ w_hh_h,  // [3H,H] f16
    const float*    __restrict__ b_hh,    // [3H]
    float*          __restrict__ gh)      // [B,3H]
{
    const int nb   = blockIdx.x;      // 0..191
    const int lane = threadIdx.x & 31;
    const int mrow = lane & 15;       // batch row for A
    const int hi   = lane >> 4;
    const int oglob = nb * 16 + mrow; // output column

    const _Float16* arow = h_in_h + (size_t)mrow * Hq;
    const _Float16* brow = w_hh_h + (size_t)oglob * Hq;

    v8f c = {};
    const int kbaseA = hi * 16;
    const int kbaseB = hi * 16;

    for (int kt = 0; kt < Hq / 32; ++kt) {
        const int k0 = kt * 32;
        v16h a, bb;
#pragma unroll
        for (int j = 0; j < 16; ++j) {
            const int ka = kbaseA + j;
            a[j]  = arow[k0 + ka];
            bb[j] = brow[k0 + kbaseB + j];
        }
        c = __builtin_amdgcn_wmma_f32_16x16x32_f16(false, a, false, bb,
                                                   (short)0, c, false, false);
    }

    const float bias = b_hh[oglob];
#pragma unroll
    for (int r = 0; r < 8; ++r) {
        const int batch = r + 8 * hi;
        gh[(size_t)batch * (3 * Hq) + oglob] = c[r] + bias;
    }
}

// ---------------------------------------------------------------------------
// Step kernel 3: GRU gates; writes hv (f32) and the f16 shadow state row t.
// gi computed on the fly (K=C=16 is tiny).
// ---------------------------------------------------------------------------
__global__ __launch_bounds__(256) void k_gru(
    const float* __restrict__ node_enc, // [B,S,C]
    const float* __restrict__ w_ih,     // [3H,C]
    const float* __restrict__ b_ih,     // [3H]
    const float* __restrict__ gh,       // [B,3H]
    const float* __restrict__ h_in,     // [B,H]
    float*       __restrict__ hv_out,   // [B,H] f32 (current hidden)
    _Float16*    __restrict__ state_h,  // [B,S,H] f16 shadow state
    int t)
{
    const int idx = blockIdx.x * blockDim.x + threadIdx.x; // 0 .. 16*1024-1
    const int b = idx >> 10;
    const int j = idx & (Hq - 1);

    const float* x = node_enc + ((size_t)b * Sq + t) * Cq;

    float gi[3];
#pragma unroll
    for (int gidx = 0; gidx < 3; ++gidx) {
        const int o = gidx * Hq + j;
        const float* wr = w_ih + (size_t)o * Cq;
        float acc = b_ih[o];
#pragma unroll
        for (int c2 = 0; c2 < Cq; ++c2) acc += x[c2] * wr[c2];
        gi[gidx] = acc;
    }

    const float* ghb = gh + (size_t)b * 3 * Hq;
    const float hr = ghb[j];
    const float hz = ghb[Hq + j];
    const float hn = ghb[2 * Hq + j];

    const float r  = sigf(gi[0] + hr);
    const float z  = sigf(gi[1] + hz);
    const float nn = tanhf(gi[2] + r * hn);
    const float h  = h_in[(size_t)b * Hq + j];
    const float hv = (1.0f - z) * nn + z * h;

    hv_out[(size_t)b * Hq + j]            = hv;
    state_h[((size_t)b * Sq + t) * Hq + j] = (_Float16)hv;
}

// ---------------------------------------------------------------------------
// Head: hv = state[:,S-1,:]; mu = hv@lin11_w.T+b; logvar = hv@lin12_w.T+b.
// Writes d_out = [hv | mu | logvar]. Grid: 64 blocks of 1 wave, 2 WMMAs each.
// ---------------------------------------------------------------------------
__global__ __launch_bounds__(32) void k_head(
    const float*    __restrict__ hv,        // [B,H] f32 (hidden at t=S-1)
    const _Float16* __restrict__ state_h,   // [B,S,H] f16
    const _Float16* __restrict__ lin11_w_h, const float* __restrict__ lin11_b,
    const _Float16* __restrict__ lin12_w_h, const float* __restrict__ lin12_b,
    float* __restrict__ out)                // [3*B*H]
{
    const int nb   = blockIdx.x;         // 0..63
    const int lane = threadIdx.x & 31;
    const int mrow = lane & 15;
    const int hi   = lane >> 4;
    const int oglob = nb * 16 + mrow;

    const _Float16* arow  = state_h + ((size_t)mrow * Sq + (Sq - 1)) * Hq;
    const _Float16* w1row = lin11_w_h + (size_t)oglob * Hq;
    const _Float16* w2row = lin12_w_h + (size_t)oglob * Hq;

    v8f c1 = {};
    v8f c2 = {};
    const int kbaseA = hi * 16;
    const int kbaseB = hi * 16;

    for (int kt = 0; kt < Hq / 32; ++kt) {
        const int k0 = kt * 32;
        v16h a, b1, b2;
#pragma unroll
        for (int j = 0; j < 16; ++j) {
            const int ka = kbaseA + j;
            a[j]  = arow[k0 + ka];
            b1[j] = w1row[k0 + kbaseB + j];
            b2[j] = w2row[k0 + kbaseB + j];
        }
        c1 = __builtin_amdgcn_wmma_f32_16x16x32_f16(false, a, false, b1,
                                                    (short)0, c1, false, false);
        c2 = __builtin_amdgcn_wmma_f32_16x16x32_f16(false, a, false, b2,
                                                    (short)0, c2, false, false);
    }

    const float bias1 = lin11_b[oglob];
    const float bias2 = lin12_b[oglob];
#pragma unroll
    for (int r = 0; r < 8; ++r) {
        const int batch = r + 8 * hi;
        out[(size_t)(Bq * Hq)     + (size_t)batch * Hq + oglob] = c1[r] + bias1; // mu
        out[(size_t)(2 * Bq * Hq) + (size_t)batch * Hq + oglob] = c2[r] + bias2; // logvar
        // hv copy-out (f32 path, full precision)
        out[(size_t)batch * Hq + oglob] = hv[(size_t)batch * Hq + oglob];
    }
}

// ---------------------------------------------------------------------------
extern "C" void kernel_launch(void* const* d_in, const int* in_sizes, int n_in,
                              void* d_out, int out_size, void* d_ws, size_t ws_size,
                              hipStream_t stream) {
    (void)in_sizes; (void)n_in; (void)out_size; (void)ws_size;

    const float* dep      = (const float*)d_in[0];   // [16,64,64]
    const float* node_enc = (const float*)d_in[1];   // [16,64,16]
    const float* gate_w   = (const float*)d_in[2];   // [1024,1024]
    const float* gate_b   = (const float*)d_in[3];   // [1024]
    const float* map_w    = (const float*)d_in[4];   // [1024,1024]
    const float* map_b    = (const float*)d_in[5];   // [1024]
    const float* w_ih     = (const float*)d_in[6];   // [3072,16]
    const float* w_hh     = (const float*)d_in[7];   // [3072,1024]
    const float* b_ih     = (const float*)d_in[8];   // [3072]
    const float* b_hh     = (const float*)d_in[9];   // [3072]
    const float* lin11_w  = (const float*)d_in[10];  // [1024,1024]
    const float* lin11_b  = (const float*)d_in[11];  // [1024]
    const float* lin12_w  = (const float*)d_in[12];  // [1024,1024]
    const float* lin12_b  = (const float*)d_in[13];  // [1024]

    float* out = (float*)d_out;

    // ---- workspace layout (f32 region then f16 region; all pow2 aligned) ----
    float* h_in = (float*)d_ws;                       // [16,1024]
    float* hv   = h_in + (size_t)Bq * Hq;             // [16,1024]
    float* gh   = hv   + (size_t)Bq * Hq;             // [16,3072]

    _Float16* state_h   = (_Float16*)(gh + (size_t)Bq * 3 * Hq); // [16,64,1024]
    _Float16* h_in_h    = state_h   + (size_t)Bq * Sq * Hq;      // [16,1024]
    _Float16* gate_w_h  = h_in_h    + (size_t)Bq * Hq;           // [H,H]
    _Float16* map_w_h   = gate_w_h  + (size_t)Hq * Hq;           // [H,H]
    _Float16* w_hh_h    = map_w_h   + (size_t)Hq * Hq;           // [3H,H]
    _Float16* lin11_w_h = w_hh_h    + (size_t)3 * Hq * Hq;       // [H,H]
    _Float16* lin12_w_h = lin11_w_h + (size_t)Hq * Hq;           // [H,H]

    // ---- one-time prep: weight conversion + f16 state zero-init ----
    const int HH = Hq * Hq;
    k_cvt<<<(HH + 255) / 256, 256, 0, stream>>>(gate_w_h, gate_w, HH);
    k_cvt<<<(HH + 255) / 256, 256, 0, stream>>>(map_w_h, map_w, HH);
    k_cvt<<<(3 * HH + 255) / 256, 256, 0, stream>>>(w_hh_h, w_hh, 3 * HH);
    k_cvt<<<(HH + 255) / 256, 256, 0, stream>>>(lin11_w_h, lin11_w, HH);
    k_cvt<<<(HH + 255) / 256, 256, 0, stream>>>(lin12_w_h, lin12_w, HH);

    const int state_elems = Bq * Sq * Hq;
    k_zero_h<<<(state_elems + 255) / 256, 256, 0, stream>>>(state_h, state_elems);

    // ---- sequential scan ----
    for (int t = 0; t < Sq; ++t) {
        dim3 g1(Hq / 16, Bq);
        k_gatemap<<<g1, 128, 0, stream>>>(dep, state_h, gate_w_h, gate_b,
                                          map_w_h, map_b, h_in, h_in_h, t);
        k_gh<<<(3 * Hq) / 16, 32, 0, stream>>>(h_in_h, w_hh_h, b_hh, gh);
        k_gru<<<(Bq * Hq) / 256, 256, 0, stream>>>(node_enc, w_ih, b_ih,
                                                   gh, h_in, hv, state_h, t);
    }

    k_head<<<Hq / 16, 32, 0, stream>>>(hv, state_h, lin11_w_h, lin11_b,
                                       lin12_w_h, lin12_b, out);
}
